// ActorNetwork_8031588844054
// MI455X (gfx1250) — matrix operand order repaired and checked by reading, not compiled, not run on hardware
//
#include <hip/hip_runtime.h>
#include <hip/hip_bf16.h>
#include <math.h>

// ---------------- problem constants ----------------
#define BS_    256
#define NUPG_  300
#define NSPG_  30
#define FIN_   32
#define C_     128
#define HH_    2
#define DD_    64
#define LL_    4
#define KK_    2
#define NU_    (BS_ * NUPG_)          // 76800
#define NS_    (BS_ * NSPG_)          // 7680
#define ND_    (BS_ * NSPG_)          // 7680
#define NTOT_  (NU_ + NS_ + ND_)      // 92160
#define ETOT_  (NU_*2 + NU_*KK_*2 + ND_ + NS_)   // 476160

typedef _Float16 v16h __attribute__((ext_vector_type(16)));
typedef _Float16 v8h  __attribute__((ext_vector_type(8)));
typedef _Float16 v2h  __attribute__((ext_vector_type(2)));
typedef float    v8f  __attribute__((ext_vector_type(8)));

__device__ __forceinline__ float geluf(float x) {
    return 0.5f * x * (1.0f + erff(x * 0.70710678118654752f));
}

__device__ __forceinline__ void atomicMaxFloat(float* addr, float val) {
    // init pattern must be -INF. Works via ordered int trick.
    if (val >= 0.0f) atomicMax((int*)addr, __float_as_int(val));
    else             atomicMin((unsigned int*)addr, __float_as_uint(val));
}

// ---------------- WMMA GEMM ----------------
// Y[M,N] = epilogue( act_in(X[M,K]) @ W[K,N] + bias )
// block: 256 threads (8 waves). tile: 128 rows x 64 cols.
// grid = (M/128, N/64). M % 128 == 0, N % 64 == 0, K % 32 == 0 (guaranteed by caller).
__global__ __launch_bounds__(256) void gemm_kernel(
    const float* __restrict__ X, int lda,
    const float* __restrict__ W, int ldw,
    const float* __restrict__ bias,
    float* __restrict__ Y, int ldy,
    int K, int inAct /*0:none 1:gelu*/, int outAct /*0:none 1:relu*/,
    const float* __restrict__ res, const float* __restrict__ gatep)
{
    __shared__ _Float16 sX[128 * 40];   // row-major, stride 40 halfs (padded)
    __shared__ _Float16 sW[64 * 40];    // col-major (col stride 40), W^T tile

    const int tid  = threadIdx.x;
    const int wid  = tid >> 5;
    const int lane = tid & 31;
    const int rowBlock = blockIdx.x * 128;
    const int colBlock = blockIdx.y * 64;

    // staging-thread geometry (loop invariant)
    const int xr  = tid >> 1;                 // X row 0..127
    const int xcb = (tid & 1) << 4;           // X col base 0 or 16
    const int wcb = (tid & 15) << 2;          // W col base 0..60 step 4
    const int wkp = (tid >> 4) << 1;          // W k pair 0..30 step 2
    const float* xpBase = X + (size_t)(rowBlock + xr) * lda + xcb;
    const float* wpBase = W + (size_t)wkp * ldw + colBlock + wcb;

    v8f acc[4] = {};

    for (int kk = 0; kk < K; kk += 32) {
        // ---- prefetch next K tile into caches while we compute this one ----
        if (kk + 32 < K) {
            __builtin_prefetch(xpBase + kk + 32, 0, 3);
            __builtin_prefetch(wpBase + (size_t)(kk + 32) * ldw, 0, 3);
        }
        // ---- stage X tile (128x32 f32 -> f16), optional gelu; vector LDS stores ----
        {
            const float* xp = xpBase + kk;
            float4 v0 = *(const float4*)(xp + 0);
            float4 v1 = *(const float4*)(xp + 4);
            float4 v2 = *(const float4*)(xp + 8);
            float4 v3 = *(const float4*)(xp + 12);
            if (inAct == 1) {
                v0.x = geluf(v0.x); v0.y = geluf(v0.y); v0.z = geluf(v0.z); v0.w = geluf(v0.w);
                v1.x = geluf(v1.x); v1.y = geluf(v1.y); v1.z = geluf(v1.z); v1.w = geluf(v1.w);
                v2.x = geluf(v2.x); v2.y = geluf(v2.y); v2.z = geluf(v2.z); v2.w = geluf(v2.w);
                v3.x = geluf(v3.x); v3.y = geluf(v3.y); v3.z = geluf(v3.z); v3.w = geluf(v3.w);
            }
            v8h h0, h1;
            h0[0] = (_Float16)v0.x; h0[1] = (_Float16)v0.y;
            h0[2] = (_Float16)v0.z; h0[3] = (_Float16)v0.w;
            h0[4] = (_Float16)v1.x; h0[5] = (_Float16)v1.y;
            h0[6] = (_Float16)v1.z; h0[7] = (_Float16)v1.w;
            h1[0] = (_Float16)v2.x; h1[1] = (_Float16)v2.y;
            h1[2] = (_Float16)v2.z; h1[3] = (_Float16)v2.w;
            h1[4] = (_Float16)v3.x; h1[5] = (_Float16)v3.y;
            h1[6] = (_Float16)v3.z; h1[7] = (_Float16)v3.w;
            _Float16* dp = &sX[xr * 40 + xcb];            // 16B aligned (80r + 32b)
            *(v8h*)(dp)     = h0;
            *(v8h*)(dp + 8) = h1;
        }
        // ---- stage W tile (32x64) transposed: read 2 K-rows coalesced,
        //      store (k,k+1) half-pairs per column as 4B LDS stores ----
        {
            const float* wp0 = wpBase + (size_t)kk * ldw;
            const float* wp1 = wp0 + ldw;
            float4 a = *(const float4*)wp0;
            float4 b = *(const float4*)wp1;
            v2h p0, p1, p2, p3;
            p0[0] = (_Float16)a.x; p0[1] = (_Float16)b.x;
            p1[0] = (_Float16)a.y; p1[1] = (_Float16)b.y;
            p2[0] = (_Float16)a.z; p2[1] = (_Float16)b.z;
            p3[0] = (_Float16)a.w; p3[1] = (_Float16)b.w;
            *(v2h*)&sW[(wcb + 0) * 40 + wkp] = p0;        // 4B aligned (80c + 2*even)
            *(v2h*)&sW[(wcb + 1) * 40 + wkp] = p1;
            *(v2h*)&sW[(wcb + 2) * 40 + wkp] = p2;
            *(v2h*)&sW[(wcb + 3) * 40 + wkp] = p3;
        }
        __syncthreads();

        // ---- build fragments and issue WMMA ----
        {
            // A fragment: lanes 0-15 -> K 0..15, lanes 16-31 -> K 16..31 (contiguous)
            const int mrow = (wid << 4) + (lane & 15);
            const int kb   = (lane >> 4) << 4;
            const _Float16* ap = &sX[mrow * 40 + kb];
            v16h afrag;
#pragma unroll
            for (int i = 0; i < 8; i++) { afrag[i] = ap[i]; afrag[8 + i] = ap[8 + i]; }

            // B fragment: lanes 0-15 -> K 0..15, lanes 16-31 -> K 16..31 (contiguous)
            const int kb2 = (lane >> 4) << 4;
#pragma unroll
            for (int nt = 0; nt < 4; nt++) {
                const _Float16* bp = &sW[((nt << 4) + (lane & 15)) * 40 + kb2];
                v16h bfrag;
#pragma unroll
                for (int i = 0; i < 16; i++) bfrag[i] = bp[i];
                acc[nt] = __builtin_amdgcn_wmma_f32_16x16x32_f16(
                    false, afrag, false, bfrag, (short)0, acc[nt], false, false);
            }
        }
        __syncthreads();
    }

    // ---- epilogue: bias, relu, gated residual blend ----
    float g = 0.0f;
    if (res) g = 1.0f / (1.0f + expf(-gatep[0]));
    const int rBase = rowBlock + (wid << 4) + ((lane >> 4) << 3);
    const int cLoc  = lane & 15;
#pragma unroll
    for (int nt = 0; nt < 4; nt++) {
        const int col = colBlock + (nt << 4) + cLoc;
        const float bv = bias ? bias[col] : 0.0f;
#pragma unroll
        for (int r = 0; r < 8; r++) {
            const int row = rBase + r;
            float v = acc[nt][r] + bv;
            if (outAct == 1) v = fmaxf(v, 0.0f);
            if (res) v = g * v + (1.0f - g) * res[(size_t)row * ldy + col];
            Y[(size_t)row * ldy + col] = v;
        }
    }
}

// ---------------- elementwise / edge kernels ----------------
__global__ void fill_kernel(float* __restrict__ p, float v, int n) {
    int t = blockIdx.x * blockDim.x + threadIdx.x;
    if (t < n) p[t] = v;
}

__global__ void finalize_amax_kernel(float* __restrict__ p, int n) {
    int t = blockIdx.x * blockDim.x + threadIdx.x;
    if (t < n) { float v = p[t]; if (!(v > -3.0e38f)) p[t] = 0.0f; }
}

__global__ void edge_logits_kernel(const int* __restrict__ src, const int* __restrict__ dst, int E,
                                   const float* __restrict__ Q, const float* __restrict__ KR,
                                   const float* __restrict__ prel,
                                   float* __restrict__ AE, float* __restrict__ AMAX, int segOff)
{
    int t = blockIdx.x * blockDim.x + threadIdx.x;
    if (t >= E * 2) return;
    int e = t >> 1, h = t & 1;
    int s = src[e], d = dst[e];
    const float4* qp = (const float4*)(Q + (size_t)d * C_ + h * DD_);
    const float4* kp = (const float4*)(KR + (size_t)s * C_ + h * DD_);
    float acc = 0.0f;
#pragma unroll
    for (int i = 0; i < 16; i++) {
        float4 a = qp[i], c = kp[i];
        acc += a.x * c.x + a.y * c.y + a.z * c.z + a.w * c.w;
    }
    float a = acc * prel[h] * 0.125f;   // 1/sqrt(64)
    AE[(size_t)e * 2 + h] = a;
    atomicMaxFloat(&AMAX[(size_t)(segOff + d) * 2 + h], a);
}

__global__ void edge_expden_kernel(const int* __restrict__ dst, int E,
                                   float* __restrict__ AE, const float* __restrict__ AMAX,
                                   float* __restrict__ DEN, int segOff)
{
    int t = blockIdx.x * blockDim.x + threadIdx.x;
    if (t >= E * 2) return;
    int e = t >> 1, h = t & 1;
    size_t si = (size_t)(segOff + dst[e]) * 2 + h;
    float ex = expf(AE[(size_t)e * 2 + h] - AMAX[si]);
    AE[(size_t)e * 2 + h] = ex;
    atomicAdd(&DEN[si], ex);
}

__global__ void edge_agg_kernel(const int* __restrict__ src, const int* __restrict__ dst, int E,
                                const float* __restrict__ VR, const float* __restrict__ AE,
                                const float* __restrict__ DEN, float* __restrict__ AGG, int segOff)
{
    int t = blockIdx.x * blockDim.x + threadIdx.x;
    if (t >= E * C_) return;
    int e = t >> 7, c = t & 127, h = c >> 6;
    int sg = segOff + dst[e];
    float ex  = AE[(size_t)e * 2 + h];
    float den = DEN[(size_t)sg * 2 + h];
    float w = ex / fmaxf(den, 1e-16f);
    atomicAdd(&AGG[(size_t)sg * C_ + c], VR[(size_t)src[e] * C_ + c] * w);
}

// out[r] = dot(X[r,:], w) + b[0]
__global__ void rowdot_kernel(const float* __restrict__ X, int ldx,
                              const float* __restrict__ w, const float* __restrict__ b,
                              float* __restrict__ out, int M, int K)
{
    int r = blockIdx.x * blockDim.x + threadIdx.x;
    if (r >= M) return;
    const float* xp = X + (size_t)r * ldx;
    float s = 0.0f;
    for (int k = 0; k < K; k += 4) {
        float4 xv = *(const float4*)(xp + k);
        float4 wv = *(const float4*)(w + k);
        s += xv.x * wv.x + xv.y * wv.y + xv.z * wv.z + xv.w * wv.w;
    }
    out[r] = s + b[0];
}

__global__ void comb_kernel(const float* __restrict__ SO, const float* __restrict__ UO,
                            const int* __restrict__ act1, float* __restrict__ COMB)
{
    int t = blockIdx.x * blockDim.x + threadIdx.x;
    if (t >= NS_ * 256) return;
    int i = t >> 8, c = t & 255;
    int b = i / NSPG_;
    float v;
    if (c < 128) v = SO[(size_t)i * 128 + c];
    else         v = UO[((size_t)b * NUPG_ + act1[b]) * 128 + (c - 128)];
    COMB[(size_t)i * 256 + c] = v;
}

// block per batch row; logits = tanh(u)*10 + mask; log_softmax, gather, entropy
__global__ __launch_bounds__(256) void softmax_head_kernel(
    const float* __restrict__ uu, const float* __restrict__ mask,
    const int* __restrict__ act, int n,
    float* __restrict__ outLogp, float* __restrict__ outEnt, int entAdd)
{
    __shared__ float red[256];
    const int b = blockIdx.x;
    const int tid = threadIdx.x;
    const float* urow = uu + (size_t)b * n;
    const float* mrow = mask + (size_t)b * n;

    float mx = -INFINITY;
    for (int j = tid; j < n; j += 256) {
        float v = tanhf(urow[j]) * 10.0f + mrow[j];
        mx = fmaxf(mx, v);
    }
    red[tid] = mx; __syncthreads();
    for (int s = 128; s > 0; s >>= 1) { if (tid < s) red[tid] = fmaxf(red[tid], red[tid + s]); __syncthreads(); }
    mx = red[0]; __syncthreads();

    float se = 0.0f, sl = 0.0f;
    for (int j = tid; j < n; j += 256) {
        float v = tanhf(urow[j]) * 10.0f + mrow[j];
        float e = expf(v - mx);
        se += e; sl += e * v;
    }
    red[tid] = se; __syncthreads();
    for (int s = 128; s > 0; s >>= 1) { if (tid < s) red[tid] += red[tid + s]; __syncthreads(); }
    se = red[0]; __syncthreads();
    red[tid] = sl; __syncthreads();
    for (int s = 128; s > 0; s >>= 1) { if (tid < s) red[tid] += red[tid + s]; __syncthreads(); }
    sl = red[0];

    if (tid == 0) {
        float logZ = mx + logf(se);
        float ent  = -(sl / se - logZ);
        int a = act[b];
        float va = tanhf(urow[a]) * 10.0f + mrow[a];
        outLogp[b] = va - logZ;
        outEnt[b]  = entAdd ? (outEnt[b] + ent) : ent;
    }
}

__global__ void acts_kernel(const int* __restrict__ a1, const int* __restrict__ a2,
                            float* __restrict__ out)
{
    int t = blockIdx.x * blockDim.x + threadIdx.x;
    if (t < BS_) { out[t] = (float)a1[t]; out[BS_ + t] = (float)a2[t]; }
}

// ---------------- host side ----------------
static inline void gemm(hipStream_t st, const float* X, int lda, const float* W, int ldw,
                        const float* bias, float* Y, int ldy, int M, int N, int K,
                        int inAct, int outAct, const float* res, const float* gate)
{
    dim3 grid(M / 128, N / 64);
    gemm_kernel<<<grid, 256, 0, st>>>(X, lda, W, ldw, bias, Y, ldy, K, inAct, outAct, res, gate);
}

extern "C" void kernel_launch(void* const* d_in, const int* in_sizes, int n_in,
                              void* d_out, int out_size, void* d_ws, size_t ws_size,
                              hipStream_t stream)
{
    (void)in_sizes; (void)n_in; (void)out_size; (void)ws_size;

    const float* xin[3] = { (const float*)d_in[0], (const float*)d_in[1], (const float*)d_in[2] };
    const float* in_w   = (const float*)d_in[3];
    const float* in_b   = (const float*)d_in[4];
    const float* k_w    = (const float*)d_in[5];
    const float* k_b    = (const float*)d_in[6];
    const float* q_w    = (const float*)d_in[7];
    const float* q_b    = (const float*)d_in[8];
    const float* v_w    = (const float*)d_in[9];
    const float* v_b    = (const float*)d_in[10];
    const float* a_rel  = (const float*)d_in[11];
    const float* m_rel  = (const float*)d_in[12];
    const float* p_rel  = (const float*)d_in[13];
    const float* o_w    = (const float*)d_in[14];
    const float* o_b    = (const float*)d_in[15];
    const float* skip   = (const float*)d_in[16];
    const float* out_w  = (const float*)d_in[17];
    const float* out_b  = (const float*)d_in[18];
    const float* fcu_w1 = (const float*)d_in[19];
    const float* fcu_b1 = (const float*)d_in[20];
    const float* fcu_w2 = (const float*)d_in[21];
    const float* fcu_b2 = (const float*)d_in[22];
    const float* fcs_w1 = (const float*)d_in[23];
    const float* fcs_b1 = (const float*)d_in[24];
    const float* fcs_w2 = (const float*)d_in[25];
    const float* fcs_b2 = (const float*)d_in[26];
    const float* mask1  = (const float*)d_in[27];
    const float* mask2  = (const float*)d_in[28];
    const int* edges[6];
    for (int r = 0; r < 6; r++) edges[r] = (const int*)d_in[29 + r];
    const int* act1 = (const int*)d_in[35];
    const int* act2 = (const int*)d_in[36];
    float* out = (float*)d_out;

    // workspace layout (floats)
    float* WSP = (float*)d_ws;
    float* XS  = WSP;                              // NTOT x 128 (node features)
    float* KB  = XS  + (size_t)NTOT_ * C_;         // NTOT x 128
    float* QB  = KB  + (size_t)NTOT_ * C_;
    float* VB  = QB  + (size_t)NTOT_ * C_;
    float* KR  = VB  + (size_t)NTOT_ * C_;         // NU x 128 (kr/vr, reused per relation)
    float* AGG = KR  + (size_t)NU_ * C_;           // NTOT x 128
    float* AE  = AGG + (size_t)NTOT_ * C_;         // ETOT x 2 (logits -> exp)
    float* AMAX = AE + (size_t)ETOT_ * 2;          // NTOT x 2
    float* DEN  = AMAX + (size_t)NTOT_ * 2;        // NTOT x 2
    // post-layer reuse:
    float* UO   = KB;   // units out proj  NU x 128
    float* SO   = QB;   // src out proj    NS x 128
    float* H1   = VB;   // fcu hidden      NU x 128
    float* UVEC = AE;   // u scalar        NU
    float* COMB = KR;   // comb            NS x 256
    float* S1   = AGG;  // fcs hidden      NS x 256
    float* SVEC = AMAX; // s scalar        NS

    static const int REL_ST[6] = {0, 1, 0, 2, 2, 1};
    static const int REL_DT[6] = {1, 0, 2, 0, 1, 2};
    const int EREL[6] = { NU_, NU_, NU_ * KK_, NU_ * KK_, ND_, NS_ };
    int aeBase[6]; { int s = 0; for (int r = 0; r < 6; r++) { aeBase[r] = s; s += EREL[r]; } }
    const int OFFT[3] = { 0, NU_, NU_ + NS_ };
    const int NT[3]   = { NU_, NS_, ND_ };

    // ---- input projection: xs[t] = relu(x @ in_w[t] + in_b[t]) ----
    for (int t = 0; t < 3; t++) {
        gemm(stream, xin[t], FIN_, in_w + (size_t)t * FIN_ * C_, C_, in_b + t * C_,
             XS + (size_t)OFFT[t] * C_, C_, NT[t], C_, FIN_, 0, 1, nullptr, nullptr);
    }

    // ---- HGT layers ----
    for (int l = 0; l < LL_; l++) {
        const float* kw = k_w + (size_t)l * 3 * C_ * C_;
        const float* qw = q_w + (size_t)l * 3 * C_ * C_;
        const float* vw = v_w + (size_t)l * 3 * C_ * C_;
        const float* kb = k_b + (size_t)l * 3 * C_;
        const float* qb = q_b + (size_t)l * 3 * C_;
        const float* vb = v_b + (size_t)l * 3 * C_;

        // init segment buffers
        fill_kernel<<<(NTOT_ * 2 + 255) / 256, 256, 0, stream>>>(AMAX, -INFINITY, NTOT_ * 2);
        fill_kernel<<<(NTOT_ * 2 + 255) / 256, 256, 0, stream>>>(DEN, 0.0f, NTOT_ * 2);
        fill_kernel<<<(NTOT_ * C_ + 255) / 256, 256, 0, stream>>>(AGG, 0.0f, NTOT_ * C_);

        // k, q, v projections
        for (int t = 0; t < 3; t++) {
            const float* x = XS + (size_t)OFFT[t] * C_;
            gemm(stream, x, C_, kw + (size_t)t * C_ * C_, C_, kb + t * C_,
                 KB + (size_t)OFFT[t] * C_, C_, NT[t], C_, C_, 0, 0, nullptr, nullptr);
            gemm(stream, x, C_, qw + (size_t)t * C_ * C_, C_, qb + t * C_,
                 QB + (size_t)OFFT[t] * C_, C_, NT[t], C_, C_, 0, 0, nullptr, nullptr);
            gemm(stream, x, C_, vw + (size_t)t * C_ * C_, C_, vb + t * C_,
                 VB + (size_t)OFFT[t] * C_, C_, NT[t], C_, C_, 0, 0, nullptr, nullptr);
        }

        // per-relation: kr = k[st] @ a_rel (per head), then edge logits + seg max
        for (int r = 0; r < 6; r++) {
            const int st = REL_ST[r], dt = REL_DT[r], E = EREL[r];
            for (int h = 0; h < HH_; h++) {
                gemm(stream, KB + (size_t)OFFT[st] * C_ + h * DD_, C_,
                     a_rel + (size_t)((l * 6 + r) * HH_ + h) * DD_ * DD_, DD_, nullptr,
                     KR + h * DD_, C_, NT[st], DD_, DD_, 0, 0, nullptr, nullptr);
            }
            edge_logits_kernel<<<(E * 2 + 255) / 256, 256, 0, stream>>>(
                edges[r], edges[r] + E, E, QB + (size_t)OFFT[dt] * C_, KR,
                p_rel + (size_t)(l * 6 + r) * HH_, AE + (size_t)aeBase[r] * 2, AMAX, OFFT[dt]);
        }

        finalize_amax_kernel<<<(NTOT_ * 2 + 255) / 256, 256, 0, stream>>>(AMAX, NTOT_ * 2);

        for (int r = 0; r < 6; r++) {
            const int dt = REL_DT[r], E = EREL[r];
            edge_expden_kernel<<<(E * 2 + 255) / 256, 256, 0, stream>>>(
                edges[r] + E, E, AE + (size_t)aeBase[r] * 2, AMAX, DEN, OFFT[dt]);
        }

        // per-relation: vr = v[st] @ m_rel (per head), weighted segment aggregation
        for (int r = 0; r < 6; r++) {
            const int st = REL_ST[r], dt = REL_DT[r], E = EREL[r];
            for (int h = 0; h < HH_; h++) {
                gemm(stream, VB + (size_t)OFFT[st] * C_ + h * DD_, C_,
                     m_rel + (size_t)((l * 6 + r) * HH_ + h) * DD_ * DD_, DD_, nullptr,
                     KR + h * DD_, C_, NT[st], DD_, DD_, 0, 0, nullptr, nullptr);
            }
            edge_agg_kernel<<<(E * C_ + 255) / 256, 256, 0, stream>>>(
                edges[r], edges[r] + E, E, KR, AE + (size_t)aeBase[r] * 2, DEN, AGG, OFFT[dt]);
        }

        // output proj with gelu-on-input + gated skip blend (in place into XS)
        for (int t = 0; t < 3; t++) {
            gemm(stream, AGG + (size_t)OFFT[t] * C_, C_,
                 o_w + (size_t)(l * 3 + t) * C_ * C_, C_, o_b + (size_t)(l * 3 + t) * C_,
                 XS + (size_t)OFFT[t] * C_, C_, NT[t], C_, C_,
                 1 /*gelu in*/, 0, XS + (size_t)OFFT[t] * C_, skip + l * 3 + t);
        }
    }

    // ---- heads ----
    // units / src output projections (type 2 not needed downstream)
    gemm(stream, XS, C_, out_w, C_, out_b, UO, C_, NU_, C_, C_, 0, 0, nullptr, nullptr);
    gemm(stream, XS + (size_t)OFFT[1] * C_, C_, out_w + (size_t)C_ * C_, C_, out_b + C_,
         SO, C_, NS_, C_, C_, 0, 0, nullptr, nullptr);

    // fcu: u = relu(UO @ w1 + b1) @ w2 + b2
    gemm(stream, UO, C_, fcu_w1, C_, fcu_b1, H1, C_, NU_, C_, C_, 0, 1, nullptr, nullptr);
    rowdot_kernel<<<(NU_ + 255) / 256, 256, 0, stream>>>(H1, C_, fcu_w2, fcu_b2, UVEC, NU_, C_);

    // head 1 softmax: logp1 -> out[512..768), ent1 -> out[1024..1280)
    softmax_head_kernel<<<BS_, 256, 0, stream>>>(UVEC, mask1, act1, NUPG_,
                                                 out + 512, out + 1024, 0);

    // comb = [SO | broadcast(UO[act1])] ; fcs
    comb_kernel<<<(NS_ * 256 + 255) / 256, 256, 0, stream>>>(SO, UO, act1, COMB);
    gemm(stream, COMB, 2 * C_, fcs_w1, 2 * C_, fcs_b1, S1, 2 * C_, NS_, 2 * C_, 2 * C_,
         0, 1, nullptr, nullptr);
    rowdot_kernel<<<(NS_ + 255) / 256, 256, 0, stream>>>(S1, 2 * C_, fcs_w2, fcs_b2, SVEC, NS_, 2 * C_);

    // head 2 softmax: logp2 -> out[768..1024), ent2 added into out[1024..1280)
    softmax_head_kernel<<<BS_, 256, 0, stream>>>(SVEC, mask2, act2, NSPG_,
                                                 out + 768, out + 1024, 1);

    // actions as floats: out[0..256)=act1, out[256..512)=act2
    acts_kernel<<<1, 256, 0, stream>>>(act1, act2, out);
}
